// GINPairV1_10024453669558
// MI455X (gfx1250) — matrix operand order repaired and checked by reading, not compiled, not run on hardware
//
#include <hip/hip_runtime.h>
#include <hip/hip_bf16.h>

typedef _Float16 half_t;
typedef __attribute__((ext_vector_type(16))) _Float16 v16h;
typedef __attribute__((ext_vector_type(8)))  float    v8f;

#define NNODES 100000
#define NEDGES 800000
#define NB     1024
#define FINF   64
#define HIDD   96
#define NCT    6        // 96/16 column tiles

// ---------------------------------------------------------------------------
// Pack a row-major f32 weight matrix W[K][96] into WMMA B-fragment order.
// Flat index: ((ct*nk + kc)*32 + lane)*16 + j
//   value = W[ kc*32 + 16*(lane>>4) + j ][ ct*16 + (lane&15) ]
// (contiguous k order used here, A packed the same; not AMD's table order: lanes 0-15 hold K=0..15, lanes 16-31 hold
//  K=16..31, two K values per VGPR)
// ---------------------------------------------------------------------------
__global__ void pack_w_kernel(const float* __restrict__ W, half_t* __restrict__ Wp,
                              int K, int Nout)
{
    int nk = K >> 5;
    int total = K * Nout;
    int t = blockIdx.x * blockDim.x + threadIdx.x;
    if (t >= total) return;
    int j  = t & 15;
    int l  = (t >> 4) & 31;
    int r  = t >> 9;
    int kc = r % nk;
    int ct = r / nk;
    int n  = ct * 16 + (l & 15);
    int k  = kc * 32 + ((l >> 4) << 4) + j;
    Wp[t] = (half_t)W[(size_t)k * Nout + n];
}

// ---------------------------------------------------------------------------
// WMMA GEMM: out[M x 96] = act( A[M x K](f32) @ W[K x 96] + bias ), f32 accum.
// Whole packed weight matrix lives in LDS (<= 36 KB). Each wave owns one
// 16-row slab and keeps SIX 16x16 f32 accumulators, sweeping all column
// tiles per A fragment -> A is read exactly once, 6 independent WMMAs issue
// back-to-back per K-chunk (no D->A/B hazard, good XDL co-execution).
// grid = ceil(M/64) blocks, block = 128 threads (4 waves).
// ---------------------------------------------------------------------------
__global__ void __launch_bounds__(128)
wmma_gemm_kernel(const float* __restrict__ A, const half_t* __restrict__ Wp,
                 const float* __restrict__ bias, float* __restrict__ out,
                 int M, int K, int lda, int ldc, int relu)
{
    __shared__ __align__(32) half_t Bs[NCT * 6 * 512];   // supports K up to 192
    const int nk = K >> 5;

    // cooperative load of ALL column tiles' B fragments into LDS
    {
        const uint4* src = (const uint4*)Wp;
        uint4* dst = (uint4*)Bs;
        const int n4 = nk * NCT * 64;                    // K*96 halves / 8
        for (int i = threadIdx.x; i < n4; i += blockDim.x)
            dst[i] = src[i];
    }
    __syncthreads();

    const int wave = threadIdx.x >> 5;
    const int lane = threadIdx.x & 31;
    const int mt   = blockIdx.x * 4 + wave;
    if (mt * 16 >= M) return;                            // wave-uniform: EXEC stays full

    const int hi = lane >> 4;                            // K-group select for A fragment
    const int m  = mt * 16 + (lane & 15);
    const float* arow = A + (size_t)m * lda;

    v8f acc[NCT];
#pragma unroll
    for (int ct = 0; ct < NCT; ++ct) acc[ct] = (v8f){};

    for (int kc = 0; kc < nk; ++kc) {
        // A fragment (16-bit A 16x32): halves 0..15 = K[kb..kb+15],
        // kb = kc*32 + 16*hi (matches packed B)
        const int kb = kc * 32 + 16 * hi;
        v16h a;
#pragma unroll
        for (int j = 0; j < 8; ++j) a[j]     = (half_t)arow[kb + j];
#pragma unroll
        for (int j = 0; j < 8; ++j) a[8 + j] = (half_t)arow[kb + 8 + j];
#pragma unroll
        for (int ct = 0; ct < NCT; ++ct) {
            const v16h b = *(const v16h*)(Bs + (size_t)((ct * nk + kc) * 32 + lane) * 16);
            acc[ct] = __builtin_amdgcn_wmma_f32_16x16x32_f16(false, a, false, b,
                                                             (short)0, acc[ct], false, false);
        }
    }

    // epilogue: C/D layout — VGPR r holds row (r + 8*hi), col = lane&15
#pragma unroll
    for (int ct = 0; ct < NCT; ++ct) {
        const int   n  = ct * 16 + (lane & 15);
        const float bv = bias[n];
#pragma unroll
        for (int r = 0; r < 8; ++r) {
            float v = acc[ct][r] + bv;
            if (relu) v = fmaxf(v, 0.0f);
            out[(size_t)(mt * 16 + r + 8 * hi) * ldc + n] = v;
        }
    }
}

// ---------------------------------------------------------------------------
// Edge aggregation: z[dst] += h[src]  (z was pre-initialized to h => z = h+agg)
// One thread handles one float4 chunk of one edge.
// ---------------------------------------------------------------------------
__global__ void agg_edges_kernel(const float* __restrict__ h, const int* __restrict__ src,
                                 const int* __restrict__ dst, float* __restrict__ z,
                                 int nE, int Fq, int ld)
{
    int t = blockIdx.x * blockDim.x + threadIdx.x;
    if (t >= nE * Fq) return;
    int e = t / Fq;
    int c = (t - e * Fq) * 4;
    const float4 v = *(const float4*)(h + (size_t)src[e] * ld + c);
    float* o = z + (size_t)dst[e] * ld + c;
    atomicAdd(o + 0, v.x);
    atomicAdd(o + 1, v.y);
    atomicAdd(o + 2, v.z);
    atomicAdd(o + 3, v.w);
}

// ---------------------------------------------------------------------------
// Mean pooling helpers
// ---------------------------------------------------------------------------
__global__ void pool_count_kernel(const int* __restrict__ batch, float* __restrict__ cnt, int n)
{
    int t = blockIdx.x * blockDim.x + threadIdx.x;
    if (t >= n) return;
    atomicAdd(&cnt[batch[t]], 1.0f);
}

__global__ void pool_sum_kernel(const float* __restrict__ h, const int* __restrict__ batch,
                                float* __restrict__ sums, int n, int Fq, int ld)
{
    int t = blockIdx.x * blockDim.x + threadIdx.x;
    if (t >= n * Fq) return;
    int i = t / Fq;
    int c = (t - i * Fq) * 4;
    const float4 v = *(const float4*)(h + (size_t)i * ld + c);
    float* o = sums + (size_t)batch[i] * HIDD + c;
    atomicAdd(o + 0, v.x);
    atomicAdd(o + 1, v.y);
    atomicAdd(o + 2, v.z);
    atomicAdd(o + 3, v.w);
}

__global__ void pool_div_kernel(const float* __restrict__ sums, const float* __restrict__ cnt,
                                float* __restrict__ hin, int side)
{
    int t = blockIdx.x * blockDim.x + threadIdx.x;
    if (t >= NB * HIDD) return;
    int b = t / HIDD;
    int f = t - b * HIDD;
    hin[(size_t)b * (2 * HIDD) + side * HIDD + f] = sums[t] / fmaxf(cnt[b], 1.0f);
}

// ---------------------------------------------------------------------------
// Final linear: out[b] = dot(relu(h[b,:]), fin_w) + fin_b ; one wave per graph
// ---------------------------------------------------------------------------
__global__ void final_lin_kernel(const float* __restrict__ h, const float* __restrict__ w,
                                 const float* __restrict__ bias, float* __restrict__ out)
{
    int b = blockIdx.x;
    int lane = threadIdx.x;
    float s = 0.0f;
    for (int i = lane; i < HIDD; i += 32)
        s += fmaxf(h[(size_t)b * HIDD + i], 0.0f) * w[i];
#pragma unroll
    for (int off = 16; off > 0; off >>= 1)
        s += __shfl_xor(s, off, 32);
    if (lane == 0) out[b] = s + bias[0];
}

// ---------------------------------------------------------------------------
extern "C" void kernel_launch(void* const* d_in, const int* in_sizes, int n_in,
                              void* d_out, int out_size, void* d_ws, size_t ws_size,
                              hipStream_t stream)
{
    const float* x_p      = (const float*)d_in[0];
    const float* x_d      = (const float*)d_in[1];
    // d_in[2], d_in[3]: edge_attr — unused by GIN (matches reference)
    const float* pw[8];  for (int i = 0; i < 8; ++i) pw[i] = (const float*)d_in[4 + i];
    const float* dw[8];  for (int i = 0; i < 8; ++i) dw[i] = (const float*)d_in[12 + i];
    const float* lin0_w = (const float*)d_in[20];  const float* lin0_b = (const float*)d_in[21];
    const float* lin1_w = (const float*)d_in[22];  const float* lin1_b = (const float*)d_in[23];
    const float* lin2_w = (const float*)d_in[24];  const float* lin2_b = (const float*)d_in[25];
    const float* fin_w  = (const float*)d_in[26];  const float* fin_b  = (const float*)d_in[27];
    const int* ei_p     = (const int*)d_in[28];
    const int* ei_d     = (const int*)d_in[29];
    const int* batch_p  = (const int*)d_in[30];
    const int* batch_d  = (const int*)d_in[31];

    // ---- workspace layout (floats first, then f16 packed weights) ----
    float* ws  = (float*)d_ws;
    float* Z   = ws;                                   // N x 96 (z = h + agg)
    float* T   = Z  + (size_t)NNODES * HIDD;           // N x 96 (hidden of MLP)
    float* Hb  = T  + (size_t)NNODES * HIDD;           // N x 96 (layer output)
    float* SUM = Hb + (size_t)NNODES * HIDD;           // B x 96
    float* CNT = SUM + (size_t)NB * HIDD;              // B
    float* HIN = CNT + NB;                             // B x 192 (concat gp|gd)
    float* HA  = HIN + (size_t)NB * 2 * HIDD;          // B x 96
    float* HB2 = HA  + (size_t)NB * HIDD;              // B x 96
    half_t* WP = (half_t*)(HB2 + (size_t)NB * HIDD);

    const int P64  = FINF * HIDD;   // 6144
    const int P96  = HIDD * HIDD;   // 9216
    const int P192 = 2 * HIDD * HIDD;
    const size_t SIDE_SZ = (size_t)P64 + 7 * (size_t)P96;
    half_t* SP[2] = { WP, WP + SIDE_SZ };
    half_t* HP    = WP + 2 * SIDE_SZ;   // lin0 | lin1 | lin2 packs

    auto packW = [&](const float* W, half_t* Wpk, int K) {
        int total = K * HIDD;
        pack_w_kernel<<<(total + 255) / 256, 256, 0, stream>>>(W, Wpk, K, HIDD);
    };

    // ---- pack all weights to WMMA fragment layout (f16) ----
    const float* const* sw[2] = { pw, dw };
    for (int s = 0; s < 2; ++s) {
        packW(sw[s][0], SP[s], FINF);                               // c0_w1 (64x96)
        packW(sw[s][2], SP[s] + P64, HIDD);                         // c0_w2 (96x96)
        for (int i = 0; i < 3; ++i) {
            packW(sw[s][4] + (size_t)i * P96, SP[s] + P64 + P96 + (size_t)i * P96, HIDD); // cw1[i]
            packW(sw[s][6] + (size_t)i * P96, SP[s] + P64 + 4 * P96 + (size_t)i * P96, HIDD); // cw2[i]
        }
    }
    packW(lin0_w, HP, 2 * HIDD);
    packW(lin1_w, HP + P192, HIDD);
    packW(lin2_w, HP + P192 + P96, HIDD);

    const dim3 gemm_blk(128);
    const dim3 gemm_grid_N((NNODES / 16 + 3) / 4);   // 1563 blocks (4 waves each)
    const dim3 gemm_grid_B((NB / 16 + 3) / 4);       // 16 blocks

    // ---- per-side GIN stack ----
    auto run_side = [&](const float* x, const int* ei, const float* const* w,
                        half_t* sp, const int* batch, int side) {
        const int* esrc = ei;
        const int* edst = ei + NEDGES;

        // layer 0: K = 64 (input features)
        hipMemcpyAsync(Z, x, (size_t)NNODES * FINF * sizeof(float),
                       hipMemcpyDeviceToDevice, stream);
        agg_edges_kernel<<<(NEDGES * (FINF / 4) + 255) / 256, 256, 0, stream>>>(
            x, esrc, edst, Z, NEDGES, FINF / 4, FINF);
        wmma_gemm_kernel<<<gemm_grid_N, gemm_blk, 0, stream>>>(
            Z, sp, w[1], T, NNODES, FINF, FINF, HIDD, 1);                 // lin1 + relu
        wmma_gemm_kernel<<<gemm_grid_N, gemm_blk, 0, stream>>>(
            T, sp + P64, w[3], Hb, NNODES, HIDD, HIDD, HIDD, 1);          // lin2, outer relu

        // layers 1..3: K = 96
        for (int i = 0; i < 3; ++i) {
            hipMemcpyAsync(Z, Hb, (size_t)NNODES * HIDD * sizeof(float),
                           hipMemcpyDeviceToDevice, stream);
            agg_edges_kernel<<<(NEDGES * (HIDD / 4) + 255) / 256, 256, 0, stream>>>(
                Hb, esrc, edst, Z, NEDGES, HIDD / 4, HIDD);
            wmma_gemm_kernel<<<gemm_grid_N, gemm_blk, 0, stream>>>(
                Z, sp + P64 + P96 + (size_t)i * P96, w[5] + (size_t)i * HIDD,
                T, NNODES, HIDD, HIDD, HIDD, 1);
            wmma_gemm_kernel<<<gemm_grid_N, gemm_blk, 0, stream>>>(
                T, sp + P64 + 4 * P96 + (size_t)i * P96, w[7] + (size_t)i * HIDD,
                Hb, NNODES, HIDD, HIDD, HIDD, (i < 2) ? 1 : 0);
        }

        // mean pool -> HIN[:, side*96 : side*96+96]
        hipMemsetAsync(SUM, 0, ((size_t)NB * HIDD + NB) * sizeof(float), stream);
        pool_count_kernel<<<(NNODES + 255) / 256, 256, 0, stream>>>(batch, CNT, NNODES);
        pool_sum_kernel<<<(NNODES * (HIDD / 4) + 255) / 256, 256, 0, stream>>>(
            Hb, batch, SUM, NNODES, HIDD / 4, HIDD);
        pool_div_kernel<<<(NB * HIDD + 255) / 256, 256, 0, stream>>>(SUM, CNT, HIN, side);
    };

    run_side(x_p, ei_p, pw, SP[0], batch_p, 0);
    run_side(x_d, ei_d, dw, SP[1], batch_d, 1);

    // ---- head MLP (B = 1024 rows, all WMMA) ----
    wmma_gemm_kernel<<<gemm_grid_B, gemm_blk, 0, stream>>>(
        HIN, HP, lin0_b, HA, NB, 2 * HIDD, 2 * HIDD, HIDD, 1);
    wmma_gemm_kernel<<<gemm_grid_B, gemm_blk, 0, stream>>>(
        HA, HP + P192, lin1_b, HB2, NB, HIDD, HIDD, HIDD, 1);
    wmma_gemm_kernel<<<gemm_grid_B, gemm_blk, 0, stream>>>(
        HB2, HP + P192 + P96, lin2_b, HA, NB, HIDD, HIDD, HIDD, 0);

    final_lin_kernel<<<NB, 32, 0, stream>>>(HA, fin_w, fin_b, (float*)d_out);
}
